// Model_53858889892017
// MI455X (gfx1250) — compile-verified, matrix operand order checked by reading, not run on hardware
//
#include <hip/hip_runtime.h>

typedef __attribute__((ext_vector_type(16))) _Float16 v16h;
typedef __attribute__((ext_vector_type(8)))  float    v8f;
typedef __attribute__((ext_vector_type(2)))  float    v2f;

#define LSTM_B 131072
#define LSTM_T 24

__device__ __forceinline__ float fsig(float v) {
  float e = __builtin_amdgcn_exp2f(-1.4426950408889634f * v);
  return __builtin_amdgcn_rcpf(1.0f + e);
}
__device__ __forceinline__ float ftanh(float v) {
  // tanh(v) = 2*sigmoid(2v) - 1
  float e = __builtin_amdgcn_exp2f(-2.8853900817779268f * v);
  return 2.0f * __builtin_amdgcn_rcpf(1.0f + e) - 1.0f;
}

__global__ __launch_bounds__(256, 1) void lstm2_wmma_kernel(
    const float* __restrict__ x,
    const float* __restrict__ w_ih0, const float* __restrict__ w_hh0,
    const float* __restrict__ b_ih0, const float* __restrict__ b_hh0,
    const float* __restrict__ w_ih1, const float* __restrict__ w_hh1,
    const float* __restrict__ b_ih1, const float* __restrict__ b_hh1,
    const float* __restrict__ wa, const float* __restrict__ ba,
    const float* __restrict__ wb, const float* __restrict__ bb,
    float* __restrict__ out)
{
  const int lane = threadIdx.x & 31;
  const int wave = threadIdx.x >> 5;
  const int half = lane >> 4;            // which 16-lane half of the wave
  const int l16  = lane & 15;
  const int batch = (blockIdx.x * 8 + wave) * 16 + l16;  // this lane's batch column

  // ---- Load weights once into WMMA A-operand layout (f16), keep in VGPRs ----
  // A 16x32 f16 layout: lanes 0-15 row M=l16, VGPR v<4: K=2v,2v+1 (+8*half),
  // v>=4: K=16+2(v-4) (+8*half).
  v16h Ahh0[8], Aih1[8], Ahh1[8];
  v2f  A0[8], A1[8];   // f32 16x4 aux A: [w_ih0 | b0 | 0 | 0], [0 | b1 | 0 | 0]
  #pragma unroll
  for (int g = 0; g < 8; ++g) {
    const int row = 16 * g + l16;
    const float* r0 = w_hh0 + row * 32;
    const float* r1 = w_ih1 + row * 32;
    const float* r2 = w_hh1 + row * 32;
    v16h a0, a1, a2;
    #pragma unroll
    for (int v = 0; v < 8; ++v) {
      const int k0 = ((v < 4) ? (2 * v) : (16 + 2 * (v - 4))) + 8 * half;
      a0[2*v] = (_Float16)r0[k0]; a0[2*v+1] = (_Float16)r0[k0+1];
      a1[2*v] = (_Float16)r1[k0]; a1[2*v+1] = (_Float16)r1[k0+1];
      a2[2*v] = (_Float16)r2[k0]; a2[2*v+1] = (_Float16)r2[k0+1];
    }
    Ahh0[g] = a0; Aih1[g] = a1; Ahh1[g] = a2;

    const float wi  = w_ih0[row];
    const float bc0 = b_ih0[row] + b_hh0[row];
    const float bc1 = b_ih1[row] + b_hh1[row];
    // f32 16x4 A layout: lanes 0-15 hold K=0,1 ; lanes 16-31 hold K=2,3 (zeros here)
    A0[g][0] = half ? 0.f : wi;   A0[g][1] = half ? 0.f : bc0;
    A1[g][0] = 0.f;               A1[g][1] = half ? 0.f : bc1;
  }

  // ---- Register-resident state ----
  float c0[16], c1[16];
  #pragma unroll
  for (int u = 0; u < 16; ++u) { c0[u] = 0.f; c1[u] = 0.f; }
  v16h h0B = {};   // h0^T in WMMA B layout (f16)
  v16h h1B = {};   // h1^T in WMMA B layout (f16)

  const float* xp = x + (long)batch * LSTM_T;

  #pragma unroll 1
  for (int t = 0; t < LSTM_T; ++t) {
    const float xt = xp[t];
    // aux B operand [x; 1; 0; 0] (f32 4x16): lanes 0-15 K=0,1; lanes 16-31 K=2,3
    v2f B0;
    B0[0] = half ? 0.f : xt;
    B0[1] = half ? 0.f : 1.f;
    const v8f zero8 = {};

    v8f gacc[8];
    // ---------------- layer 0: gates0^T = [w_ih0|b0]@[x;1] + W_hh0 @ h0^T ----
    #pragma unroll
    for (int g = 0; g < 8; ++g) {
      v8f acc = __builtin_amdgcn_wmma_f32_16x16x4_f32(
          false, A0[g], false, B0, (short)0, zero8, false, false);
      gacc[g] = __builtin_amdgcn_wmma_f32_16x16x32_f16(
          false, Ahh0[g], false, h0B, (short)0, acc, false, false);
    }
    // cell 0 (lane owns 16 hidden units; i/f/g/o live in tiles g, g+2, g+4, g+6)
    v16h nh0;
    #pragma unroll
    for (int u = 0; u < 16; ++u) {
      const int g = u >> 3, r = u & 7;
      const float iv = fsig (gacc[g    ][r]);
      const float fv = fsig (gacc[g + 2][r]);
      const float gv = ftanh(gacc[g + 4][r]);
      const float ov = fsig (gacc[g + 6][r]);
      const float cn = fv * c0[u] + iv * gv;
      c0[u] = cn;
      nh0[u] = (_Float16)(ov * ftanh(cn));   // lands exactly in next B-layout slot
    }
    h0B = nh0;

    // ---------------- layer 1: gates1^T = [0|b1]@[x;1] + W_ih1@h0^T + W_hh1@h1^T
    #pragma unroll
    for (int g = 0; g < 8; ++g) {
      v8f acc = __builtin_amdgcn_wmma_f32_16x16x4_f32(
          false, A1[g], false, B0, (short)0, zero8, false, false);
      acc = __builtin_amdgcn_wmma_f32_16x16x32_f16(
          false, Aih1[g], false, h0B, (short)0, acc, false, false);
      gacc[g] = __builtin_amdgcn_wmma_f32_16x16x32_f16(
          false, Ahh1[g], false, h1B, (short)0, acc, false, false);
    }
    v16h nh1;
    #pragma unroll
    for (int u = 0; u < 16; ++u) {
      const int g = u >> 3, r = u & 7;
      const float iv = fsig (gacc[g    ][r]);
      const float fv = fsig (gacc[g + 2][r]);
      const float gv = ftanh(gacc[g + 4][r]);
      const float ov = fsig (gacc[g + 6][r]);
      const float cn = fv * c1[u] + iv * gv;
      c1[u] = cn;
      nh1[u] = (_Float16)(ov * ftanh(cn));
    }
    h1B = nh1;
  }

  // ---- head: 3 * (tanh(h1 @ wa^T + ba) @ wb^T + bb) ----
  // Lane owns hidden j = (u&7) + ((u>>3)<<4) + 8*half; partner half via shfl_xor(16).
  float partial[8];
  #pragma unroll
  for (int a = 0; a < 8; ++a) {
    float acc = 0.f;
    #pragma unroll
    for (int u = 0; u < 16; ++u) {
      const int j = (u & 7) + ((u >> 3) << 4) + 8 * half;
      acc += (float)h1B[u] * wa[a * 32 + j];
    }
    partial[a] = acc;
  }
  float res = 0.f;
  #pragma unroll
  for (int a = 0; a < 8; ++a) {
    float s = partial[a] + __shfl_xor(partial[a], 16, 32);
    res += ftanh(s + ba[a]) * wb[a];
  }
  res = 3.0f * (res + bb[0]);
  if (half == 0) out[batch] = res;
}

extern "C" void kernel_launch(void* const* d_in, const int* in_sizes, int n_in,
                              void* d_out, int out_size, void* d_ws, size_t ws_size,
                              hipStream_t stream) {
  const float* x     = (const float*)d_in[0];
  const float* w_ih0 = (const float*)d_in[1];
  const float* w_hh0 = (const float*)d_in[2];
  const float* b_ih0 = (const float*)d_in[3];
  const float* b_hh0 = (const float*)d_in[4];
  const float* w_ih1 = (const float*)d_in[5];
  const float* w_hh1 = (const float*)d_in[6];
  const float* b_ih1 = (const float*)d_in[7];
  const float* b_hh1 = (const float*)d_in[8];
  const float* wa    = (const float*)d_in[9];
  const float* ba    = (const float*)d_in[10];
  const float* wb    = (const float*)d_in[11];
  const float* bb    = (const float*)d_in[12];
  float* out = (float*)d_out;

  dim3 block(256);                       // 8 waves, 16 batch columns each
  dim3 grid(LSTM_B / 128);               // 1024 blocks -> 8192 waves total
  hipLaunchKernelGGL(lstm2_wmma_kernel, grid, block, 0, stream,
                     x, w_ih0, w_hh0, b_ih0, b_hh0,
                     w_ih1, w_hh1, b_ih1, b_hh1,
                     wa, ba, wb, bb, out);
}
